// RWKV_Tmix_x070_28209345200467
// MI455X (gfx1250) — compile-verified, matrix operand order checked by reading, not run on hardware
//
#include <hip/hip_runtime.h>

typedef __attribute__((ext_vector_type(16))) __bf16 v16bf;
typedef __attribute__((ext_vector_type(8)))  float  v8f;

__device__ __forceinline__ unsigned short f2bf(float f) {
    unsigned int u = __float_as_uint(f);
    unsigned int r = (u + 0x7FFFu + ((u >> 16) & 1u)) >> 16;
    return (unsigned short)r;
}

union FragU {
    v16bf v;
    uint4 q[2];
};

// ---------------------------------------------------------------------------
// Generic tiled WMMA GEMM:  Out[M,N] = epi( A[M,K](bf16) x W , bias )
//   BTRANS=1: W is [N,K] fp32 (computes A @ W^T)   (big attention weights)
//   BTRANS=0: W is [K,N] fp32                      (LoRA weights)
// Block tile 128x128, 256 threads (8 waves, 4x2), wave tile 32x64, K-step 32.
// A tile staged via GLOBAL_LOAD_ASYNC_TO_LDS_B128 (ASYNCcnt path, bf16 direct);
// B tile staged through VGPRs with fp32->bf16 conversion, stored [n][k].
// EPI: 0 ident->f32   1 tanh->bf16   2 bias+sigmoid->f32
//      3 sigmoid->bf16 4 bias+decay(exp(-exp(-softplus(-s)-0.5)))->f32
//      5 ident->bf16
// ---------------------------------------------------------------------------
template <bool BTRANS, int EPI>
__global__ __launch_bounds__(256) void gemm_wmma(
    const unsigned short* __restrict__ A, const float* __restrict__ W,
    void* __restrict__ Out, const float* __restrict__ bias, int M, int N, int K)
{
    __shared__ unsigned short ash[128][48];   // [m][k], row 96B (16B aligned)
    __shared__ unsigned short bsh[128][48];   // [n][k]

    const int tid  = threadIdx.x;
    const int wv   = tid >> 5;
    const int lane = tid & 31;
    const int half = lane >> 4;
    const int r16  = lane & 15;
    const int wm   = wv & 3;          // M sub-tile (32 rows)
    const int wn   = wv >> 2;         // N sub-tile (64 cols)

    const int m0 = blockIdx.y * 128;
    const int n0 = blockIdx.x * 128;

    v8f acc[2][4];
#pragma unroll
    for (int i = 0; i < 2; ++i)
#pragma unroll
        for (int j = 0; j < 4; ++j)
#pragma unroll
            for (int g = 0; g < 8; ++g) acc[i][j][g] = 0.f;

    for (int k0 = 0; k0 < K; k0 += 32) {
        // ---- A tile 128x32 bf16: async global->LDS (no VGPR roundtrip) ----
#pragma unroll
        for (int c = 0; c < 2; ++c) {
            int chunk = tid + c * 256;            // 512 chunks of 8 bf16
            int row   = chunk >> 2;
            int kc    = (chunk & 3) * 8;
            unsigned long long ga =
                (unsigned long long)(A + (size_t)(m0 + row) * K + k0 + kc);
            unsigned la = (unsigned)(unsigned long long)&ash[row][kc];
            asm volatile("global_load_async_to_lds_b128 %0, %1, off"
                         :: "v"(la), "v"(ga) : "memory");
        }
        // ---- B tile as [n][k] bf16, fp32->bf16 on the fly ----
        if (BTRANS) {
#pragma unroll
            for (int c = 0; c < 4; ++c) {
                int chunk = tid + c * 256;        // 1024 chunks of 4 floats
                int n     = chunk >> 3;
                int kc    = (chunk & 7) * 4;
                float4 f  = make_float4(0.f, 0.f, 0.f, 0.f);
                if (n0 + n < N)
                    f = *(const float4*)(W + (size_t)(n0 + n) * K + k0 + kc);
                unsigned short* p = &bsh[n][kc];
                p[0] = f2bf(f.x); p[1] = f2bf(f.y);
                p[2] = f2bf(f.z); p[3] = f2bf(f.w);
            }
        } else {
#pragma unroll
            for (int c = 0; c < 16; ++c) {
                int e  = tid + c * 256;           // 4096 scalars
                int kk = e >> 7;
                int n  = e & 127;
                float f = (n0 + n < N) ? W[(size_t)(k0 + kk) * N + n0 + n] : 0.f;
                bsh[n][kk] = f2bf(f);
            }
        }
        asm volatile("s_wait_asynccnt 0x0" ::: "memory");
        __syncthreads();

        // ---- fragments + 8 WMMAs (32x64 per wave) ----
        FragU af[2];
#pragma unroll
        for (int i = 0; i < 2; ++i) {
            int row = wm * 32 + i * 16 + r16;
            af[i].q[0] = *(const uint4*)(&ash[row][half * 8]);
            af[i].q[1] = *(const uint4*)(&ash[row][16 + half * 8]);
        }
#pragma unroll
        for (int j = 0; j < 4; ++j) {
            FragU bfm;
            int n = wn * 64 + j * 16 + r16;
            bfm.q[0] = *(const uint4*)(&bsh[n][half * 8]);
            bfm.q[1] = *(const uint4*)(&bsh[n][16 + half * 8]);
#pragma unroll
            for (int i = 0; i < 2; ++i)
                acc[i][j] = __builtin_amdgcn_wmma_f32_16x16x32_bf16(
                    false, af[i].v, false, bfm.v, (short)0, acc[i][j], false, false);
        }
        __syncthreads();
    }

    // ---- epilogue: lane holds (M = g + 8*half, N = r16) of each 16x16 tile
#pragma unroll
    for (int i = 0; i < 2; ++i)
#pragma unroll
    for (int j = 0; j < 4; ++j) {
        int n = n0 + wn * 64 + j * 16 + r16;
        if (n >= N) continue;
        float bval = 0.f;
        if (EPI == 2 || EPI == 4) bval = bias[n];
#pragma unroll
        for (int g = 0; g < 8; ++g) {
            int    m   = m0 + wm * 32 + i * 16 + half * 8 + g;
            float  va  = acc[i][j][g];
            size_t idx = (size_t)m * N + n;
            if (EPI == 0) {
                ((float*)Out)[idx] = va;
            } else if (EPI == 1) {
                ((unsigned short*)Out)[idx] = f2bf(tanhf(va));
            } else if (EPI == 2) {
                float s = va + bval;
                ((float*)Out)[idx] = 1.f / (1.f + __expf(-s));
            } else if (EPI == 3) {
                ((unsigned short*)Out)[idx] = f2bf(1.f / (1.f + __expf(-va)));
            } else if (EPI == 4) {
                float s = va + bval;
                float w = -log1pf(__expf(-s)) - 0.5f;    // log-decay
                ((float*)Out)[idx] = __expf(-__expf(w)); // decay factor
            } else {
                ((unsigned short*)Out)[idx] = f2bf(va);
            }
        }
    }
}

// ---------------------------------------------------------------------------
// Time-shift + six lerp blends -> bf16 GEMM operands
// ---------------------------------------------------------------------------
__global__ __launch_bounds__(256) void blend_kernel(
    const float* __restrict__ x,
    const float* __restrict__ mr, const float* __restrict__ mw,
    const float* __restrict__ mk, const float* __restrict__ mv,
    const float* __restrict__ ma, const float* __restrict__ mg,
    unsigned short* __restrict__ xr, unsigned short* __restrict__ xw,
    unsigned short* __restrict__ xk, unsigned short* __restrict__ xv,
    unsigned short* __restrict__ xa, unsigned short* __restrict__ xg,
    int T, int C)
{
    int row = blockIdx.x;          // b*T + t
    int t   = row % T;
    size_t base = (size_t)row * C;
    for (int c = threadIdx.x; c < C; c += 256) {
        float xc = x[base + c];
        float xp = (t > 0) ? x[base - C + c] : 0.f;
        float xx = xp - xc;
        xr[base + c] = f2bf(xc + xx * mr[c]);
        xw[base + c] = f2bf(xc + xx * mw[c]);
        xk[base + c] = f2bf(xc + xx * mk[c]);
        xv[base + c] = f2bf(xc + xx * mv[c]);
        xa[base + c] = f2bf(xc + xx * ma[c]);
        xg[base + c] = f2bf(xc + xx * mg[c]);
    }
}

// ---------------------------------------------------------------------------
// Per-head prep: kk = normalize(k*k_k); k *= (1+(a-1)*k_a); v = lerp(v, v_first)
// one wave32 per (b,t,h) row of 64; lane covers j and j+32
// ---------------------------------------------------------------------------
__global__ __launch_bounds__(256) void prep_kernel(
    float* __restrict__ k, const float* __restrict__ a,
    const float* __restrict__ vsig, float* __restrict__ v,
    const float* __restrict__ v_first, const float* __restrict__ k_k,
    const float* __restrict__ k_a, float* __restrict__ kkout, int rows)
{
    int wv = threadIdx.x >> 5, lane = threadIdx.x & 31;
    int rh = blockIdx.x * 8 + wv;
    if (rh >= rows) return;
    int h = rh & 31;                       // H = 32
    size_t base = (size_t)rh * 64;
    int c0 = h * 64 + lane, c1 = c0 + 32;

    float k0 = k[base + lane],      k1 = k[base + lane + 32];
    float kk0 = k0 * k_k[c0],       kk1 = k1 * k_k[c1];
    float ss = kk0 * kk0 + kk1 * kk1;
#pragma unroll
    for (int m = 16; m; m >>= 1) ss += __shfl_xor(ss, m, 32);
    float inv = 1.f / fmaxf(sqrtf(ss), 1e-12f);
    kkout[base + lane]      = kk0 * inv;
    kkout[base + lane + 32] = kk1 * inv;

    float a0 = a[base + lane], a1 = a[base + lane + 32];
    k[base + lane]      = k0 * (1.f + (a0 - 1.f) * k_a[c0]);
    k[base + lane + 32] = k1 * (1.f + (a1 - 1.f) * k_a[c1]);

    float v0 = v[base + lane], v1 = v[base + lane + 32];
    float s0 = vsig[base + lane], s1 = vsig[base + lane + 32];
    v[base + lane]      = v0 + (v_first[base + lane]      - v0) * s0;
    v[base + lane + 32] = v1 + (v_first[base + lane + 32] - v1) * s1;
}

// ---------------------------------------------------------------------------
// Sequential RWKV7 scan: one block of 64 threads per (b,h); state row / thread
// ---------------------------------------------------------------------------
__global__ __launch_bounds__(64) void scan_kernel(
    const float* __restrict__ r, const float* __restrict__ decay,
    const float* __restrict__ k, const float* __restrict__ v,
    const float* __restrict__ kk, const float* __restrict__ a,
    float* __restrict__ y, int T, int H)
{
    __shared__ float sr[64], sw[64], sk[64], sv[64], sz[64], sb[64];
    int bh = blockIdx.x, b = bh / H, h = bh % H;
    int i  = threadIdx.x;
    float S[64];
#pragma unroll
    for (int j = 0; j < 64; ++j) S[j] = 0.f;

    for (int t = 0; t < T; ++t) {
        size_t base = ((size_t)(b * T + t) * H + h) * 64;
        sr[i] = r[base + i];
        sw[i] = decay[base + i];
        sk[i] = k[base + i];
        sv[i] = v[base + i];
        float kkv = kk[base + i], av = a[base + i];
        sz[i] = -kkv;
        sb[i] = kkv * av;
        __syncthreads();

        float sa = 0.f;
#pragma unroll
        for (int j = 0; j < 64; ++j) sa += S[j] * sz[j];
        float vi = sv[i], yv = 0.f;
#pragma unroll
        for (int j = 0; j < 64; ++j) {
            S[j] = S[j] * sw[j] + sa * sb[j] + vi * sk[j];
            yv  += S[j] * sr[j];
        }
        y[base + i] = yv;
        __syncthreads();
    }
}

// ---------------------------------------------------------------------------
// GroupNorm (per head) + bonus + gate -> bf16 operand for output projection
// ---------------------------------------------------------------------------
__global__ __launch_bounds__(256) void post_kernel(
    const float* __restrict__ y, const float* __restrict__ r,
    const float* __restrict__ k, const float* __restrict__ v,
    const float* __restrict__ g, const float* __restrict__ r_k,
    const float* __restrict__ gn_w, const float* __restrict__ gn_b,
    unsigned short* __restrict__ u, int rows, float eps)
{
    int wv = threadIdx.x >> 5, lane = threadIdx.x & 31;
    int rh = blockIdx.x * 8 + wv;
    if (rh >= rows) return;
    int h = rh & 31;
    size_t base = (size_t)rh * 64;
    int c0 = h * 64 + lane, c1 = c0 + 32;

    float y0 = y[base + lane], y1 = y[base + lane + 32];
    float s = y0 + y1, s2 = y0 * y0 + y1 * y1;
#pragma unroll
    for (int m = 16; m; m >>= 1) { s += __shfl_xor(s, m, 32); s2 += __shfl_xor(s2, m, 32); }
    float mu  = s * (1.f / 64.f);
    float var = s2 * (1.f / 64.f) - mu * mu;
    float inv = rsqrtf(var + eps);

    float dot = r[base + lane]      * k[base + lane]      * r_k[c0]
              + r[base + lane + 32] * k[base + lane + 32] * r_k[c1];
#pragma unroll
    for (int m = 16; m; m >>= 1) dot += __shfl_xor(dot, m, 32);

    float u0 = ((y0 - mu) * inv * gn_w[c0] + gn_b[c0] + dot * v[base + lane])      * g[base + lane];
    float u1 = ((y1 - mu) * inv * gn_w[c1] + gn_b[c1] + dot * v[base + lane + 32]) * g[base + lane + 32];
    u[base + lane]      = f2bf(u0);
    u[base + lane + 32] = f2bf(u1);
}

// ---------------------------------------------------------------------------
extern "C" void kernel_launch(void* const* d_in, const int* in_sizes, int n_in,
                              void* d_out, int out_size, void* d_ws, size_t ws_size,
                              hipStream_t stream)
{
    const int B = 2, T = 2048, C = 2048, H = 32;
    const int M = B * T;
    const float EPS_GN = 1e-5f * 64.f;

    const float* x       = (const float*)d_in[0];
    const float* v_first = (const float*)d_in[1];
    const float* x_r = (const float*)d_in[2];
    const float* x_w = (const float*)d_in[3];
    const float* x_k = (const float*)d_in[4];
    const float* x_v = (const float*)d_in[5];
    const float* x_a = (const float*)d_in[6];
    const float* x_g = (const float*)d_in[7];
    const float* w0  = (const float*)d_in[8];
    const float* w1  = (const float*)d_in[9];
    const float* w2  = (const float*)d_in[10];
    const float* a0  = (const float*)d_in[11];
    const float* a1  = (const float*)d_in[12];
    const float* a2  = (const float*)d_in[13];
    const float* v0  = (const float*)d_in[14];
    const float* v1  = (const float*)d_in[15];
    const float* v2  = (const float*)d_in[16];
    const float* g1  = (const float*)d_in[17];
    const float* g2  = (const float*)d_in[18];
    const float* k_k = (const float*)d_in[19];
    const float* k_a = (const float*)d_in[20];
    const float* r_k = (const float*)d_in[21];
    const float* W_r = (const float*)d_in[22];
    const float* W_k = (const float*)d_in[23];
    const float* W_v = (const float*)d_in[24];
    const float* W_o = (const float*)d_in[25];
    const float* gn_w = (const float*)d_in[26];
    const float* gn_b = (const float*)d_in[27];
    (void)in_sizes; (void)n_in; (void)out_size; (void)ws_size;

    // ---- workspace bump allocator ----
    char* wp = (char*)d_ws;
    auto alloc = [&](size_t bytes) {
        char* p = wp;
        wp += (bytes + 255) & ~(size_t)255;
        return p;
    };
    const size_t MC = (size_t)M * C;
    unsigned short* xr = (unsigned short*)alloc(MC * 2);
    unsigned short* xw = (unsigned short*)alloc(MC * 2);
    unsigned short* xk = (unsigned short*)alloc(MC * 2);
    unsigned short* xv = (unsigned short*)alloc(MC * 2);
    unsigned short* xa = (unsigned short*)alloc(MC * 2);
    unsigned short* xg = (unsigned short*)alloc(MC * 2);
    float* rbuf  = (float*)alloc(MC * 4);
    float* kbuf  = (float*)alloc(MC * 4);
    float* vbuf  = (float*)alloc(MC * 4);
    float* dbuf  = (float*)alloc(MC * 4);   // decay
    float* abuf  = (float*)alloc(MC * 4);
    float* vsig  = (float*)alloc(MC * 4);
    float* gbuf  = (float*)alloc(MC * 4);
    float* kkbuf = (float*)alloc(MC * 4);
    float* ybuf  = (float*)alloc(MC * 4);
    unsigned short* hw = (unsigned short*)alloc((size_t)M * 64 * 2);
    unsigned short* ha = (unsigned short*)alloc((size_t)M * 64 * 2);
    unsigned short* hv = (unsigned short*)alloc((size_t)M * 32 * 2);
    unsigned short* hg = (unsigned short*)alloc((size_t)M * 128 * 2);
    unsigned short* ub = (unsigned short*)alloc(MC * 2);

    dim3 blk(256);
    auto grid_of = [&](int N_) { return dim3((unsigned)((N_ + 127) / 128), (unsigned)(M / 128)); };

    // 1) blends
    blend_kernel<<<M, 256, 0, stream>>>(x, x_r, x_w, x_k, x_v, x_a, x_g,
                                        xr, xw, xk, xv, xa, xg, T, C);

    // 2) big projections  r/k/v = x? @ W.T
    gemm_wmma<true, 0><<<grid_of(C), blk, 0, stream>>>(xr, W_r, rbuf, nullptr, M, C, C);
    gemm_wmma<true, 0><<<grid_of(C), blk, 0, stream>>>(xk, W_k, kbuf, nullptr, M, C, C);
    gemm_wmma<true, 0><<<grid_of(C), blk, 0, stream>>>(xv, W_v, vbuf, nullptr, M, C, C);

    // 3) LoRA down-projections (fused activations)
    gemm_wmma<false, 1><<<grid_of(64),  blk, 0, stream>>>(xw, w1, hw, nullptr, M, 64,  C); // tanh
    gemm_wmma<false, 5><<<grid_of(64),  blk, 0, stream>>>(xa, a1, ha, nullptr, M, 64,  C);
    gemm_wmma<false, 5><<<grid_of(32),  blk, 0, stream>>>(xv, v1, hv, nullptr, M, 32,  C);
    gemm_wmma<false, 3><<<grid_of(128), blk, 0, stream>>>(xg, g1, hg, nullptr, M, 128, C); // sigmoid

    // 4) LoRA up-projections (fused bias + nonlinearity)
    gemm_wmma<false, 4><<<grid_of(C), blk, 0, stream>>>(hw, w2, dbuf, w0, M, C, 64);  // decay
    gemm_wmma<false, 2><<<grid_of(C), blk, 0, stream>>>(ha, a2, abuf, a0, M, C, 64);  // sigmoid(a0+..)
    gemm_wmma<false, 2><<<grid_of(C), blk, 0, stream>>>(hv, v2, vsig, v0, M, C, 32);  // sigmoid(v0+..)
    gemm_wmma<false, 0><<<grid_of(C), blk, 0, stream>>>(hg, g2, gbuf, nullptr, M, C, 128);

    // 5) per-head prep (kk-normalize, k-scale, v-blend)
    int rows = M * H;
    prep_kernel<<<rows / 8, 256, 0, stream>>>(kbuf, abuf, vsig, vbuf, v_first,
                                              k_k, k_a, kkbuf, rows);

    // 6) sequential scan
    scan_kernel<<<B * H, 64, 0, stream>>>(rbuf, dbuf, kbuf, vbuf, kkbuf, abuf,
                                          ybuf, T, H);

    // 7) groupnorm + bonus + gate
    post_kernel<<<rows / 8, 256, 0, stream>>>(ybuf, rbuf, kbuf, vbuf, gbuf, r_k,
                                              gn_w, gn_b, ub, rows, EPS_GN);

    // 8) output projection -> d_out[0 : M*C]
    gemm_wmma<true, 0><<<grid_of(C), blk, 0, stream>>>(ub, W_o, (float*)d_out,
                                                       nullptr, M, C, C);

    // 9) second tuple output: v_first passthrough
    hipMemcpyAsync((float*)d_out + MC, v_first, MC * sizeof(float),
                   hipMemcpyDeviceToDevice, stream);
}
